// Attention_57243324121446
// MI455X (gfx1250) — matrix operand order repaired and checked by reading, not compiled, not run on hardware
//
#include <hip/hip_runtime.h>
#include <hip/hip_bf16.h>

// ---------------------------------------------------------------- types
typedef __bf16 bf16;
typedef __attribute__((ext_vector_type(16))) __bf16 v16bf;
typedef __attribute__((ext_vector_type(8)))  __bf16 v8bf;
typedef __attribute__((ext_vector_type(8)))  float  v8f;

union V16 { v16bf v; v8bf h[2]; };

#define B_SZ     2
#define N_SEQ    2048
#define D_MODEL  1536
#define N_HEADS  8
#define DQK      128
#define DV       192
#define NPAIR    512
#define DPAIR    128
#define QKV_COLS 1344   // 8*128 + 128 + 192
#define SCALE_F  0.08838834764831845f  // 128^-0.5

__device__ __forceinline__ bf16 f2bf(float f) {
  unsigned u = __builtin_bit_cast(unsigned, f);
  unsigned r = u + 0x7fffu + ((u >> 16) & 1u);
  unsigned short s = (unsigned short)(r >> 16);
  return __builtin_bit_cast(bf16, s);
}

__device__ __forceinline__ v8f wmma_bf16(v16bf a, v16bf b, v8f c) {
  // D = A(16x32 bf16) * B(32x16 bf16) + C(16x16 f32)
  return __builtin_amdgcn_wmma_f32_16x16x32_bf16(false, a, false, b,
                                                 (short)0, c, false, false);
}

// ---------------------------------------------------------------- converts
__global__ __launch_bounds__(256) void af_cvt_bf16(const float* __restrict__ src,
                                                   bf16* __restrict__ dst, size_t n) {
  size_t i = (size_t)blockIdx.x * 256 + threadIdx.x;
  if (i < n) dst[i] = f2bf(src[i]);
}

// dst[n][k] = src[k][n]  (src: K rows x N cols) -> bf16 transposed weights
__global__ __launch_bounds__(256) void af_transpose_bf16(const float* __restrict__ src,
                                                         bf16* __restrict__ dst,
                                                         int K, int N) {
  size_t i = (size_t)blockIdx.x * 256 + threadIdx.x;
  if (i >= (size_t)K * N) return;
  int n = (int)(i / K), k = (int)(i % K);
  dst[i] = f2bf(src[(size_t)k * N + n]);
}

// ---------------------------------------------------------------- generic WMMA GEMM
// C[M,N] = A[M,K](bf16) * BT[N,K](bf16)^T  (+ bias[N]),  M%64==0, N%64==0, K%32==0
__global__ __launch_bounds__(128) void af_gemm_bf16(const bf16* __restrict__ A,
                                                    const bf16* __restrict__ BT,
                                                    const float* __restrict__ bias,
                                                    float* __restrict__ C,
                                                    int M, int N, int K) {
  const int wave = threadIdx.x >> 5, lane = threadIdx.x & 31;
  const int half = lane >> 4, ln = lane & 15;
  const int m0 = blockIdx.y * 64 + wave * 16;
  const int n0 = blockIdx.x * 64;
  (void)M;

  v8f acc[4];
#pragma unroll
  for (int nt = 0; nt < 4; ++nt)
#pragma unroll
    for (int r = 0; r < 8; ++r) acc[nt][r] = 0.f;

  const bf16* arow = A + (size_t)(m0 + ln) * K;
  for (int kk = 0; kk < K; kk += 32) {
    V16 af;
    af.h[0] = *(const v8bf*)(arow + kk + half * 16);
    af.h[1] = *(const v8bf*)(arow + kk + half * 16 + 8);
#pragma unroll
    for (int nt = 0; nt < 4; ++nt) {
      const bf16* bp = BT + (size_t)(n0 + nt * 16 + ln) * K + kk + half * 16;
      v16bf bfv = *(const v16bf*)bp;
      acc[nt] = wmma_bf16(af.v, bfv, acc[nt]);
    }
  }
#pragma unroll
  for (int nt = 0; nt < 4; ++nt) {
    int col = n0 + nt * 16 + ln;
    float badd = bias ? bias[col] : 0.f;
#pragma unroll
    for (int r = 0; r < 8; ++r) {
      int row = m0 + r + 8 * half;
      C[(size_t)row * N + col] = acc[nt][r] + badd;
    }
  }
}

// ---------------------------------------------------------------- layernorm -> bf16
template <int DLEN>
__device__ __forceinline__ void ln_row(const float* __restrict__ src,
                                       const float* __restrict__ w,
                                       const float* __restrict__ bb,
                                       bf16* __restrict__ dst, int lane) {
  constexpr int NEL = DLEN / 32;
  float x[NEL];
  float sum = 0.f;
#pragma unroll
  for (int i = 0; i < NEL; ++i) { x[i] = src[lane + i * 32]; sum += x[i]; }
#pragma unroll
  for (int o = 16; o >= 1; o >>= 1) sum += __shfl_xor(sum, o, 32);
  float mu = sum * (1.0f / DLEN);
  float var = 0.f;
#pragma unroll
  for (int i = 0; i < NEL; ++i) { float d = x[i] - mu; var += d * d; }
#pragma unroll
  for (int o = 16; o >= 1; o >>= 1) var += __shfl_xor(var, o, 32);
  var *= (1.0f / DLEN);
  float rs = rsqrtf(var + 1e-5f);
#pragma unroll
  for (int i = 0; i < NEL; ++i) {
    int d = lane + i * 32;
    dst[d] = f2bf((x[i] - mu) * rs * w[d] + bb[d]);
  }
}

__global__ __launch_bounds__(256) void af_ln_kernel(
    const float* __restrict__ qkv,
    const float* __restrict__ qw, const float* __restrict__ qb,
    const float* __restrict__ kw, const float* __restrict__ kb,
    const float* __restrict__ vw, const float* __restrict__ vb,
    bf16* __restrict__ q_bf, bf16* __restrict__ k_bf, bf16* __restrict__ v_bf) {
  const int wid = blockIdx.x * 8 + (threadIdx.x >> 5);
  const int lane = threadIdx.x & 31;
  const int NQ = B_SZ * N_SEQ * N_HEADS;            // 32768
  if (wid < NQ) {
    int b = wid / (N_SEQ * N_HEADS);
    int rem = wid % (N_SEQ * N_HEADS);
    int n = rem / N_HEADS, h = rem % N_HEADS;
    const float* src = qkv + (size_t)(b * N_SEQ + n) * QKV_COLS + h * DQK;
    bf16* dst = q_bf + ((size_t)(b * N_HEADS + h) * N_SEQ + n) * DQK;
    ln_row<DQK>(src, qw, qb, dst, lane);
  } else if (wid < NQ + B_SZ * N_SEQ) {
    int r = wid - NQ;
    const float* src = qkv + (size_t)r * QKV_COLS + N_HEADS * DQK;
    ln_row<DQK>(src, kw, kb, k_bf + (size_t)r * DQK, lane);
  } else {
    int r = wid - NQ - B_SZ * N_SEQ;
    const float* src = qkv + (size_t)r * QKV_COLS + N_HEADS * DQK + DQK;
    ln_row<DV>(src, vw, vb, v_bf + (size_t)r * DV, lane);
  }
}

// ---------------------------------------------------------------- pairwise variance
__global__ __launch_bounds__(256) void af_pair_var(const float* __restrict__ p,
                                                   float* __restrict__ sums) {
  __shared__ float s_sum[DPAIR], s_sq[DPAIR];
  size_t tid = (size_t)blockIdx.x * 256 + threadIdx.x;
  const size_t total = (size_t)B_SZ * NPAIR * NPAIR * DPAIR;
  const size_t stride = (size_t)gridDim.x * 256;    // multiple of 128
  float s = 0.f, q = 0.f;
  for (size_t i = tid; i < total; i += stride) { float x = p[i]; s += x; q += x * x; }
  if (threadIdx.x < DPAIR) { s_sum[threadIdx.x] = 0.f; s_sq[threadIdx.x] = 0.f; }
  __syncthreads();
  int c = (int)(tid % DPAIR);
  atomicAdd(&s_sum[c], s);
  atomicAdd(&s_sq[c], q);
  __syncthreads();
  if (threadIdx.x < DPAIR) {
    atomicAdd(&sums[threadIdx.x], s_sum[threadIdx.x]);
    atomicAdd(&sums[DPAIR + threadIdx.x], s_sq[threadIdx.x]);
  }
}

__global__ void af_stats_final(const float* __restrict__ sums,
                               const float* __restrict__ gamma,
                               const float* __restrict__ rvar,
                               float* __restrict__ scale) {
  int c = threadIdx.x;
  if (c >= DPAIR) return;
  const float cnt = (float)((size_t)B_SZ * NPAIR * NPAIR);
  float mean = sums[c] / cnt;
  float var = sums[DPAIR + c] / cnt - mean * mean;
  float rv = rvar[c] * 0.9f + 0.1f * var;
  scale[c] = gamma[c] * rsqrtf(rv + 1e-5f);
}

// ---------------------------------------------------------------- coarse bias (b,i,j,h)
__global__ __launch_bounds__(256) void af_pair_bias(const float* __restrict__ p,
                                                    const float* __restrict__ scale,
                                                    const float* __restrict__ beta,
                                                    const float* __restrict__ wb,
                                                    float* __restrict__ coarse) {
  const int wid = blockIdx.x * 8 + (threadIdx.x >> 5);   // (b*512+i)*512+j
  const int lane = threadIdx.x & 31;
  const float* row = p + (size_t)wid * DPAIR;
  float partial[N_HEADS];
#pragma unroll
  for (int h = 0; h < N_HEADS; ++h) partial[h] = 0.f;
#pragma unroll
  for (int k = 0; k < 4; ++k) {
    int c = lane + k * 32;
    float x = row[c] * scale[c] + beta[c];
    float g = 0.5f * x * (1.f + erff(x * 0.70710678f));   // exact GELU
#pragma unroll
    for (int h = 0; h < N_HEADS; ++h) partial[h] += g * wb[c * N_HEADS + h];
  }
#pragma unroll
  for (int h = 0; h < N_HEADS; ++h)
#pragma unroll
    for (int o = 16; o >= 1; o >>= 1) partial[h] += __shfl_xor(partial[h], o, 32);
  if (lane < N_HEADS) coarse[(size_t)wid * N_HEADS + lane] = partial[lane];
}

// ---------------------------------------------------------------- flash attention
// grid (32 m-tiles, 8 heads, 2 batch), block 128 (4 waves, 16 rows each)
// Double-buffered through VGPRs: next K/V tile loads are issued right after the
// barrier and retire while the wave runs 8 S-WMMAs + softmax + 12 PV-WMMAs.
__global__ __launch_bounds__(128) void af_flash(const bf16* __restrict__ qb,
                                                const bf16* __restrict__ kb,
                                                const bf16* __restrict__ vb,
                                                const float* __restrict__ coarse,
                                                bf16* __restrict__ obuf) {
  __shared__ __align__(32) bf16 s_k[32 * DQK];      // 8 KB   K tile (row-major)
  __shared__ __align__(32) bf16 s_vt[DV * 32];      // 12 KB  V tile transposed [dv][key]
  __shared__ __align__(16) bf16 s_p[4 * 16 * 32];   // 4 KB   P staging per wave

  const int b = blockIdx.z, h = blockIdx.y, mt = blockIdx.x;
  const int wave = threadIdx.x >> 5, lane = threadIdx.x & 31;
  const int half = lane >> 4, ln = lane & 15;
  const int m0 = mt * 64 + wave * 16;

  // Q fragments: 16 rows x 128 features, 4 K-chunks of 32
  const bf16* qrow = qb + ((size_t)(b * N_HEADS + h) * N_SEQ + m0 + ln) * DQK;
  V16 qf[4];
#pragma unroll
  for (int kk = 0; kk < 4; ++kk) {
    qf[kk].h[0] = *(const v8bf*)(qrow + kk * 32 + half * 16);
    qf[kk].h[1] = *(const v8bf*)(qrow + kk * 32 + half * 16 + 8);
  }

  v8f acc[12];
#pragma unroll
  for (int nt = 0; nt < 12; ++nt)
#pragma unroll
    for (int r = 0; r < 8; ++r) acc[nt][r] = 0.f;
  float m_i[8], l_i[8];
#pragma unroll
  for (int r = 0; r < 8; ++r) { m_i[r] = -1e30f; l_i[r] = 0.f; }

  // register-staged tiles: K tile = 512 uint4 (16/key-row), V tile = 768 uint4 (24/key-row)
  const uint4* ksrc = (const uint4*)(kb + (size_t)b * N_SEQ * DQK);
  const uint4* vsrc = (const uint4*)(vb + (size_t)b * N_SEQ * DV);
  uint4 kreg[4], vreg[6];
#pragma unroll
  for (int i = 0; i < 4; ++i) kreg[i] = ksrc[threadIdx.x + i * 128];
#pragma unroll
  for (int i = 0; i < 6; ++i) vreg[i] = vsrc[threadIdx.x + i * 128];

  for (int j0 = 0; j0 < N_SEQ; j0 += 32) {
    // commit staged registers to LDS
    {
      uint4* dk = (uint4*)s_k;
#pragma unroll
      for (int i = 0; i < 4; ++i) dk[threadIdx.x + i * 128] = kreg[i];
#pragma unroll
      for (int i = 0; i < 6; ++i) {
        int f = threadIdx.x + i * 128;   // uint4 index within 32x192 tile
        int key = f / 24;                // 24 uint4 per key row
        int dv0 = (f % 24) * 8;
        bf16 tmp[8];
        *(uint4*)tmp = vreg[i];
#pragma unroll
        for (int e = 0; e < 8; ++e) s_vt[(dv0 + e) * 32 + key] = tmp[e];
      }
    }
    __syncthreads();

    // issue next tile's global loads now; they overlap all compute below
    if (j0 + 32 < N_SEQ) {
      size_t kof = (size_t)(j0 + 32) * 16;
      size_t vof = (size_t)(j0 + 32) * 24;
#pragma unroll
      for (int i = 0; i < 4; ++i) kreg[i] = ksrc[kof + threadIdx.x + i * 128];
#pragma unroll
      for (int i = 0; i < 6; ++i) vreg[i] = vsrc[vof + threadIdx.x + i * 128];
    }

    // S = Q K^T for a 16x32 strip; scale + coarse bias + softclamp
    float sv[2][8];
#pragma unroll
    for (int t = 0; t < 2; ++t) {
      v8f c;
#pragma unroll
      for (int r = 0; r < 8; ++r) c[r] = 0.f;
#pragma unroll
      for (int kk = 0; kk < 4; ++kk) {
        v16bf bfv = *(const v16bf*)(s_k + (t * 16 + ln) * DQK + kk * 32 + half * 16);
        c = wmma_bf16(qf[kk].v, bfv, c);
      }
      int jg = (j0 + t * 16 + ln) >> 2;             // 4x key upsample
#pragma unroll
      for (int r = 0; r < 8; ++r) {
        int ig = (m0 + r + 8 * half) >> 2;          // 4x query upsample
        float x = c[r] * SCALE_F +
                  coarse[(((size_t)b * NPAIR + ig) * NPAIR + jg) * N_HEADS + h];
        sv[t][r] = tanhf(x * 0.2f) * 5.0f;          // softclamp 5
      }
    }

    // online softmax over the 32-key strip (row groups of 16 lanes)
    float alpha[8];
#pragma unroll
    for (int r = 0; r < 8; ++r) {
      float mx = fmaxf(sv[0][r], sv[1][r]);
#pragma unroll
      for (int o = 8; o >= 1; o >>= 1) mx = fmaxf(mx, __shfl_xor(mx, o, 16));
      float mnew = fmaxf(m_i[r], mx);
      float a = __expf(m_i[r] - mnew);
      float p0 = __expf(sv[0][r] - mnew);
      float p1 = __expf(sv[1][r] - mnew);
      sv[0][r] = p0; sv[1][r] = p1;
      float sum = p0 + p1;
#pragma unroll
      for (int o = 8; o >= 1; o >>= 1) sum += __shfl_xor(sum, o, 16);
      l_i[r] = l_i[r] * a + sum;
      m_i[r] = mnew;
      alpha[r] = a;
    }
#pragma unroll
    for (int nt = 0; nt < 12; ++nt)
#pragma unroll
      for (int r = 0; r < 8; ++r) acc[nt][r] *= alpha[r];

    // D-layout -> A-layout for P via wave-private LDS (same-wave DS ops are
    // in-order per the ISA, so no block barrier is needed here)
    bf16* pw = s_p + wave * (16 * 32);
#pragma unroll
    for (int t = 0; t < 2; ++t)
#pragma unroll
      for (int r = 0; r < 8; ++r)
        pw[(r + 8 * half) * 32 + t * 16 + ln] = f2bf(sv[t][r]);

    V16 pf;
    pf.h[0] = *(const v8bf*)(pw + ln * 32 + half * 16);
    pf.h[1] = *(const v8bf*)(pw + ln * 32 + half * 16 + 8);

    // O += P(16x32) @ V(32x192)
#pragma unroll
    for (int nt = 0; nt < 12; ++nt) {
      v16bf vf = *(const v16bf*)(s_vt + (nt * 16 + ln) * 32 + half * 16);
      acc[nt] = wmma_bf16(pf.v, vf, acc[nt]);
    }
    __syncthreads();   // all waves done with s_k/s_vt before next commit
  }

  float inv[8];
#pragma unroll
  for (int r = 0; r < 8; ++r) inv[r] = 1.f / l_i[r];
#pragma unroll
  for (int nt = 0; nt < 12; ++nt)
#pragma unroll
    for (int r = 0; r < 8; ++r) {
      int row = m0 + r + 8 * half;
      obuf[((size_t)b * N_SEQ + row) * (N_HEADS * DV) + h * DV + nt * 16 + ln] =
          f2bf(acc[nt][r] * inv[r]);
    }
}

// ---------------------------------------------------------------- launcher
extern "C" void kernel_launch(void* const* d_in, const int* in_sizes, int n_in,
                              void* d_out, int out_size, void* d_ws, size_t ws_size,
                              hipStream_t stream) {
  (void)in_sizes; (void)n_in; (void)out_size; (void)ws_size;
  const float* x       = (const float*)d_in[0];
  const float* pair    = (const float*)d_in[1];
  const float* w_qkv   = (const float*)d_in[2];
  const float* q_w     = (const float*)d_in[3];
  const float* q_b     = (const float*)d_in[4];
  const float* k_w     = (const float*)d_in[5];
  const float* k_b     = (const float*)d_in[6];
  const float* v_w     = (const float*)d_in[7];
  const float* v_b     = (const float*)d_in[8];
  const float* gamma   = (const float*)d_in[9];
  const float* beta    = (const float*)d_in[10];
  const float* rvar    = (const float*)d_in[11];
  const float* w_bias  = (const float*)d_in[12];
  const float* w_out   = (const float*)d_in[13];
  const float* b_out   = (const float*)d_in[14];
  float* out = (float*)d_out;

  char* ws = (char*)d_ws;
  size_t off = 0;
  auto alloc = [&](size_t bytes) -> void* {
    void* p = ws + off;
    off += (bytes + 255) & ~(size_t)255;
    return p;
  };
  const size_t ROWS = (size_t)B_SZ * N_SEQ;               // 4096
  bf16*  xbf    = (bf16*)alloc(ROWS * D_MODEL * 2);
  bf16*  wqkvT  = (bf16*)alloc((size_t)QKV_COLS * D_MODEL * 2);
  float* qkvf   = (float*)alloc(ROWS * QKV_COLS * 4);
  bf16*  q_bf   = (bf16*)alloc((size_t)B_SZ * N_HEADS * N_SEQ * DQK * 2);
  bf16*  k_bf   = (bf16*)alloc((size_t)B_SZ * N_SEQ * DQK * 2);
  bf16*  v_bf   = (bf16*)alloc((size_t)B_SZ * N_SEQ * DV * 2);
  float* stats  = (float*)alloc(3 * DPAIR * 4);           // sum, sumsq, scale
  float* coarse = (float*)alloc((size_t)B_SZ * NPAIR * NPAIR * N_HEADS * 4);
  bf16*  obuf   = (bf16*)alloc(ROWS * (N_HEADS * DV) * 2);
  bf16*  woutT  = (bf16*)alloc((size_t)D_MODEL * D_MODEL * 2);
  float* scale  = stats + 2 * DPAIR;

  // 1) bf16 conversions of activations and (transposed) weights
  {
    size_t n = ROWS * D_MODEL;
    af_cvt_bf16<<<(unsigned)((n + 255) / 256), 256, 0, stream>>>(x, xbf, n);
  }
  {
    size_t n = (size_t)QKV_COLS * D_MODEL;
    af_transpose_bf16<<<(unsigned)((n + 255) / 256), 256, 0, stream>>>(
        w_qkv, wqkvT, D_MODEL, QKV_COLS);
  }
  {
    size_t n = (size_t)D_MODEL * D_MODEL;
    af_transpose_bf16<<<(unsigned)((n + 255) / 256), 256, 0, stream>>>(
        w_out, woutT, D_MODEL, D_MODEL);
  }

  // 2) QKV projection (WMMA):  [4096,1536] x [1536,1344]
  af_gemm_bf16<<<dim3(QKV_COLS / 64, (unsigned)(ROWS / 64)), 128, 0, stream>>>(
      xbf, wqkvT, nullptr, qkvf, (int)ROWS, QKV_COLS, D_MODEL);

  // 3) per-head layernorms -> bf16 q/k/v
  af_ln_kernel<<<(B_SZ * N_SEQ * (N_HEADS + 2)) / 8, 256, 0, stream>>>(
      qkvf, q_w, q_b, k_w, k_b, v_w, v_b, q_bf, k_bf, v_bf);

  // 4) pairwise BatchRMSNorm stats (two-pass) + GELU-bias projection
  hipMemsetAsync(stats, 0, 2 * DPAIR * sizeof(float), stream);
  af_pair_var<<<2048, 256, 0, stream>>>(pair, stats);
  af_stats_final<<<1, 128, 0, stream>>>(stats, gamma, rvar, scale);
  af_pair_bias<<<(B_SZ * NPAIR * NPAIR) / 8, 256, 0, stream>>>(
      pair, scale, beta, w_bias, coarse);

  // 5) flash attention with coarse bias + softclamp (WMMA, double-buffered)
  af_flash<<<dim3(N_SEQ / 64, N_HEADS, B_SZ), 128, 0, stream>>>(
      q_bf, k_bf, v_bf, coarse, obuf);

  // 6) output projection (WMMA): [4096,1536] x [1536,1536] + b_out
  af_gemm_bf16<<<dim3(D_MODEL / 64, (unsigned)(ROWS / 64)), 128, 0, stream>>>(
      obuf, woutT, b_out, out, (int)ROWS, D_MODEL, D_MODEL);
}
